// SudokuIREDTrainer_35167192219715
// MI455X (gfx1250) — compile-verified, matrix operand order checked by reading, not run on hardware
//
#include <hip/hip_runtime.h>
#include <hip/hip_bf16.h>
#include <math.h>

// ---------------- problem constants (from reference) ----------------
#define B_SZ      16384
#define K_SCHED   50
#define H_DIM     2048
#define EMB_DIM   128
#define C_CH      7
#define CELLS     36            // G*G
#define YN        252           // C*G*G
#define FEAT_DIM  416           // YN + CELLS + EMB  (13*32, WMMA-friendly)
#define SIG_MAX   0.95f
#define SIG_MIN   0.01f
#define EPS_PROB  0.1f
#define NEG_SCALE 0.3f
#define REFINE_STEPS 5
#define REFINE_LR 0.15f

typedef __bf16 bf16_t;
typedef __bf16 bf16x16 __attribute__((ext_vector_type(16)));
typedef __bf16 bf16x8  __attribute__((ext_vector_type(8)));
typedef float  f32x8   __attribute__((ext_vector_type(8)));

// ---------------- device helpers ----------------
__device__ __forceinline__ float sigma_of(int k) {
  // sigmas[k] = SIG_MIN + (SIG_MAX-SIG_MIN)*cos^2(k/(K-1) * pi/2)   (schedule reversed)
  float ang = ((float)k / (float)(K_SCHED - 1)) * 1.5707963267948966f;
  float c = __cosf(ang);
  return SIG_MIN + (SIG_MAX - SIG_MIN) * c * c;
}

__device__ __forceinline__ float gelu_f(float x) {
  float u = 0.7978845608028654f * (x + 0.044715f * x * x * x);
  return 0.5f * x * (1.0f + tanhf(u));
}
__device__ __forceinline__ float gelu_grad_f(float x) {
  float x2 = x * x;
  float u = 0.7978845608028654f * (x + 0.044715f * x2 * x);
  float t = tanhf(u);
  float du = 0.7978845608028654f * (1.0f + 0.134145f * x2);
  return 0.5f * (1.0f + t) + 0.5f * x * (1.0f - t * t) * du;
}
__device__ __forceinline__ float clip01(float v) { return fminf(fmaxf(v, 0.0f), 1.0f); }

__device__ __forceinline__ f32x8 wmma_bf16(const bf16x16& a, const bf16x16& b, f32x8 c) {
  return __builtin_amdgcn_wmma_f32_16x16x32_bf16(false, a, false, b, (short)0, c,
                                                 false, false);
}

// ---------------- WMMA GEMM:  C[M,N] = A[M,K] * Bt[N,K]^T  ----------------
// Register-blocked: each wave computes a 32x32 output tile = 2x2 WMMA tiles,
// reusing each A fragment across 2 B tiles (4 wmma per 8 b128 loads).
// Block = 8 waves stacked along M -> block tile 256(M) x 32(N); N granularity 32
// keeps FEAT_DIM=416 (13x32) exactly tileable.
// Fragment layout per CDNA5 ISA 7.12.2 (16-bit A 16x32):
//   lane = row (l&15); elems 0..7 -> K = k0 + 8*(l>>4)..+7,
//   elems 8..15 -> K = k0+16+8*(l>>4)..+7  (two b128 loads per fragment).
//   D (f32 16x16): elem e -> row = e + 8*(l>>4), col = l&15.
enum { EPI_FWD = 0, EPI_DGELU = 1, EPI_PLAIN = 2 };

template <int EPI>
__global__ __launch_bounds__(256)
void k_wmma_gemm(const bf16_t* __restrict__ A,
                 const bf16_t* __restrict__ Bt,
                 const float*  __restrict__ bias,   // [N]      (EPI_FWD)
                 const float*  __restrict__ dpre,   // [M,N]    (EPI_DGELU: * gelu'(dpre))
                 float*        __restrict__ out_pre,// [M,N]    (EPI_FWD)
                 bf16_t*       __restrict__ out_bf, // [M,N]    (EPI_FWD act / EPI_DGELU grad)
                 float*        __restrict__ out_f32,// [M,N]    (EPI_PLAIN)
                 int M, int K, int N)
{
  (void)M;
  const int lane = threadIdx.x & 31;
  const int wave = threadIdx.x >> 5;              // 8 waves along M
  const int hl   = lane >> 4;                     // half-lane group 0/1
  const int l15  = lane & 15;
  const int m0   = blockIdx.y * 256 + wave * 32;  // wave's 32x32 tile origin
  const int n0   = blockIdx.x * 32;
  const int koff = hl << 3;                       // 0 or 8

  const bf16_t* ap0 = A  + (size_t)(m0 + l15) * K + koff;
  const bf16_t* ap1 = ap0 + (size_t)16 * K;
  const bf16_t* bp0 = Bt + (size_t)(n0 + l15) * K + koff;
  const bf16_t* bp1 = bp0 + (size_t)16 * K;

  union Frag { bf16x16 v; bf16x8 h[2]; };
  f32x8 acc[2][2] = {};

  for (int k0 = 0; k0 < K; k0 += 32) {
    Frag a0, a1, b0, b1;
    a0.h[0] = *(const bf16x8*)(ap0 + k0);
    a0.h[1] = *(const bf16x8*)(ap0 + k0 + 16);
    a1.h[0] = *(const bf16x8*)(ap1 + k0);
    a1.h[1] = *(const bf16x8*)(ap1 + k0 + 16);
    b0.h[0] = *(const bf16x8*)(bp0 + k0);
    b0.h[1] = *(const bf16x8*)(bp0 + k0 + 16);
    b1.h[0] = *(const bf16x8*)(bp1 + k0);
    b1.h[1] = *(const bf16x8*)(bp1 + k0 + 16);
    // speculative prefetch of the next K tiles (global_prefetch_b8)
    __builtin_prefetch(ap0 + k0 + 96, 0, 1);
    __builtin_prefetch(ap1 + k0 + 96, 0, 1);
    __builtin_prefetch(bp0 + k0 + 96, 0, 1);
    __builtin_prefetch(bp1 + k0 + 96, 0, 1);
    acc[0][0] = wmma_bf16(a0.v, b0.v, acc[0][0]);
    acc[0][1] = wmma_bf16(a0.v, b1.v, acc[0][1]);
    acc[1][0] = wmma_bf16(a1.v, b0.v, acc[1][0]);
    acc[1][1] = wmma_bf16(a1.v, b1.v, acc[1][1]);
  }

#pragma unroll
  for (int im = 0; im < 2; ++im) {
#pragma unroll
    for (int in = 0; in < 2; ++in) {
      const int ncol = n0 + in * 16 + l15;
#pragma unroll
      for (int e = 0; e < 8; ++e) {
        int mrow = m0 + im * 16 + e + (hl << 3);
        size_t idx = (size_t)mrow * N + ncol;
        float v = acc[im][in][e];
        if constexpr (EPI == EPI_FWD) {
          v += bias[ncol];
          out_pre[idx] = v;
          out_bf[idx]  = (bf16_t)gelu_f(v);
        } else if constexpr (EPI == EPI_DGELU) {
          out_bf[idx] = (bf16_t)(v * gelu_grad_f(dpre[idx]));
        } else {
          out_f32[idx] = v;
        }
      }
    }
  }
}

// ---------------- weight conversion: f32 -> bf16 (natural + transposed) ----------------
__global__ void k_cvt(const float* __restrict__ src, bf16_t* __restrict__ dn,
                      bf16_t* __restrict__ dt, int rows, int cols)
{
  size_t n = (size_t)rows * cols;
  for (size_t t = (size_t)blockIdx.x * blockDim.x + threadIdx.x; t < n;
       t += (size_t)gridDim.x * blockDim.x) {
    int r = (int)(t / cols), c = (int)(t % cols);
    bf16_t v = (bf16_t)src[t];
    dn[t] = v;
    dt[(size_t)c * rows + r] = v;
  }
}

// ---------------- probability-noise kernel (one thread per sample*cell) ----------------
__global__ void k_noise(const float* __restrict__ y, const float* __restrict__ eps,
                        const int* __restrict__ k_idx, float* __restrict__ yhat)
{
  int t = blockIdx.x * blockDim.x + threadIdx.x;
  if (t >= B_SZ * CELLS) return;
  int i = t / CELLS, g = t % CELLS;
  float sig = sigma_of(k_idx[i]);
  float sq = sqrtf(fmaxf(1.0f - sig * sig, 1e-8f));
  const float* yr = y   + (size_t)i * YN;
  const float* er = eps + (size_t)i * YN;
  float yh[C_CH];
#pragma unroll
  for (int c = 0; c < C_CH; ++c)
    yh[c] = clip01(sq * yr[c * CELLS + g] + sig * (er[c * CELLS + g] * EPS_PROB));
  bool filled = yh[0] > 0.5f;
  float vs = 0.0f;
#pragma unroll
  for (int c = 1; c < C_CH; ++c) vs += yh[c];
  float inv = 1.0f / (vs + 1e-8f);
  float* o = yhat + (size_t)i * YN;
  o[g] = yh[0];
#pragma unroll
  for (int c = 1; c < C_CH; ++c)
    o[c * CELLS + g] = clip01(filled ? yh[c] * inv : yh[c]);
}

// ---------------- feature assembly: [clip(y)(252) | cond(36) | k_emb(128)] -> bf16 ----------------
__global__ void k_feat(const float* __restrict__ yv, const float* __restrict__ cond,
                       const int* __restrict__ k_idx, const float* __restrict__ k_emb,
                       bf16_t* __restrict__ feat)
{
  size_t n = (size_t)B_SZ * FEAT_DIM;
  for (size_t t = (size_t)blockIdx.x * blockDim.x + threadIdx.x; t < n;
       t += (size_t)gridDim.x * blockDim.x) {
    int i = (int)(t / FEAT_DIM), j = (int)(t % FEAT_DIM);
    float v;
    if (j < YN)                 v = clip01(yv[(size_t)i * YN + j]);
    else if (j < YN + CELLS)    v = cond[(size_t)i * CELLS + (j - YN)];
    else                        v = k_emb[(size_t)k_idx[i] * EMB_DIM + (j - YN - CELLS)];
    feat[t] = (bf16_t)v;
  }
}

// ---------------- e = h2 . W3 + b3 (one wave per sample) ----------------
__global__ __launch_bounds__(256)
void k_dot_w3(const bf16_t* __restrict__ h2, const float* __restrict__ W3,
              const float* __restrict__ b3, float* __restrict__ e)
{
  int w = blockIdx.x * 8 + (threadIdx.x >> 5);
  int lane = threadIdx.x & 31;
  if (w >= B_SZ) return;
  const bf16_t* row = h2 + (size_t)w * H_DIM;
  float s = 0.0f;
  for (int j = lane; j < H_DIM; j += 32) s += (float)row[j] * W3[j];
  for (int off = 16; off; off >>= 1) s += __shfl_xor(s, off, 32);
  if (lane == 0) e[w] = s + b3[0];
}

// ---------------- dh2 = W3[j] * gelu'(pre2)  (dE/de == 1) ----------------
__global__ void k_dh2(const float* __restrict__ pre2, const float* __restrict__ W3,
                      bf16_t* __restrict__ dh2)
{
  size_t n = (size_t)B_SZ * H_DIM;
  for (size_t t = (size_t)blockIdx.x * blockDim.x + threadIdx.x; t < n;
       t += (size_t)gridDim.x * blockDim.x) {
    int j = (int)(t % H_DIM);
    dh2[t] = (bf16_t)(W3[j] * gelu_grad_f(pre2[t]));
  }
}

// ---------------- MSE accumulation ----------------
__global__ void k_mse(const float* __restrict__ dfeat, const float* __restrict__ yhat,
                      const float* __restrict__ y_true, const int* __restrict__ k_idx,
                      float* __restrict__ acc)
{
  size_t n = (size_t)B_SZ * YN;
  float local = 0.0f;
  for (size_t t = (size_t)blockIdx.x * blockDim.x + threadIdx.x; t < n;
       t += (size_t)gridDim.x * blockDim.x) {
    int i = (int)(t / YN), j = (int)(t % YN);
    float yh = yhat[t];
    float score = dfeat[(size_t)i * FEAT_DIM + j];
    if (!(yh >= 0.0f && yh <= 1.0f)) score = 0.0f;  // clip gradient mask
    float tgt = (yh - y_true[t]) / (sigma_of(k_idx[i]) + 1e-8f);
    float d = score - tgt;
    local += d * d;
  }
  for (int off = 16; off; off >>= 1) local += __shfl_xor(local, off, 32);
  if ((threadIdx.x & 31) == 0) atomicAdd(&acc[0], local);
}

// ---------------- negative construction (sort 36 scores, corrupt) ----------------
__global__ void k_make_neg(const float* __restrict__ y_true, const float* __restrict__ sel,
                           const float* __restrict__ rv, float* __restrict__ yneg)
{
  int i = blockIdx.x * blockDim.x + threadIdx.x;
  if (i >= B_SZ) return;
  const float* y = y_true + (size_t)i * YN;
  const float* s = sel + (size_t)i * CELLS;
  const float* r = rv + (size_t)i * (C_CH - 1) * CELLS;
  float sc[CELLS];
  int nf = 0;
  for (int g = 0; g < CELLS; ++g) {
    bool f = y[g] > 0.5f;
    nf += f ? 1 : 0;
    sc[g] = f ? s[g] : 2.0f;
  }
  int num_c = (int)floorf((float)nf * NEG_SCALE);
  float srt[CELLS];
  for (int g = 0; g < CELLS; ++g) srt[g] = sc[g];
  for (int a = 1; a < CELLS; ++a) {           // ascending insertion sort
    float v = srt[a]; int b = a - 1;
    while (b >= 0 && srt[b] > v) { srt[b + 1] = srt[b]; --b; }
    srt[b + 1] = v;
  }
  int idx = num_c - 1; if (idx < 0) idx = 0;
  float thr = (num_c > 0) ? srt[idx] : -1.0f;
  float* o = yneg + (size_t)i * YN;
  for (int g = 0; g < CELLS; ++g) {
    bool filled = y[g] > 0.5f;
    bool corrupt = filled && (sc[g] <= thr);
    float vs = 0.0f;
    for (int c = 0; c < C_CH - 1; ++c) vs += r[c * CELLS + g];
    float inv = 1.0f / vs;
    o[g] = corrupt ? 0.0f : y[g];
    for (int c = 0; c < C_CH - 1; ++c)
      o[(c + 1) * CELLS + g] = corrupt ? r[c * CELLS + g] * inv : y[(c + 1) * CELLS + g];
  }
}

// ---------------- refine update: y += LR * dE/dy, clip ----------------
__global__ void k_refine(const float* __restrict__ dfeat, float* __restrict__ yneg)
{
  size_t n = (size_t)B_SZ * YN;
  for (size_t t = (size_t)blockIdx.x * blockDim.x + threadIdx.x; t < n;
       t += (size_t)gridDim.x * blockDim.x) {
    int i = (int)(t / YN), j = (int)(t % YN);
    float y = yneg[t];
    float g = dfeat[(size_t)i * FEAT_DIM + j];
    if (!(y >= 0.0f && y <= 1.0f)) g = 0.0f;
    yneg[t] = clip01(y + REFINE_LR * g);
  }
}

// ---------------- contrastive + reg accumulation ----------------
__global__ void k_ereduce(const float* __restrict__ ep, const float* __restrict__ en,
                          float* __restrict__ acc)
{
  float lc = 0.0f, lr = 0.0f;
  for (int i = blockIdx.x * blockDim.x + threadIdx.x; i < B_SZ;
       i += gridDim.x * blockDim.x) {
    float a = ep[i], b = en[i];
    float mn = fminf(a, b);
    // -log_softmax(-[a,b])[0] = a + log(exp(-a)+exp(-b))
    lc += a - mn + logf(__expf(mn - a) + __expf(mn - b));
    lr += a + b;
  }
  for (int off = 16; off; off >>= 1) {
    lc += __shfl_xor(lc, off, 32);
    lr += __shfl_xor(lr, off, 32);
  }
  if ((threadIdx.x & 31) == 0) {
    atomicAdd(&acc[1], lc);
    atomicAdd(&acc[2], lr);
  }
}

__global__ void k_zero(float* acc) { if (threadIdx.x < 8) acc[threadIdx.x] = 0.0f; }

__global__ void k_final(const float* __restrict__ acc, float* __restrict__ out)
{
  if (threadIdx.x == 0 && blockIdx.x == 0) {
    float mse = acc[0] / (float)((size_t)B_SZ * YN);
    float con = acc[1] / (float)B_SZ;
    float reg = acc[2] / (float)B_SZ;
    out[0] = 1.0f * mse + 1.0f * con + 0.01f * reg;
  }
}

// ---------------- host orchestration ----------------
extern "C" void kernel_launch(void* const* d_in, const int* in_sizes, int n_in,
                              void* d_out, int out_size, void* d_ws, size_t ws_size,
                              hipStream_t stream)
{
  (void)in_sizes; (void)n_in; (void)out_size; (void)ws_size;
  const float* y_true = (const float*)d_in[0];
  const float* cond   = (const float*)d_in[1];
  const int*   k_idx  = (const int*)d_in[2];
  const float* eps_p  = (const float*)d_in[3];
  const float* eps_n  = (const float*)d_in[4];
  const float* sel    = (const float*)d_in[5];
  const float* rv     = (const float*)d_in[6];
  const float* W1     = (const float*)d_in[7];
  const float* b1     = (const float*)d_in[8];
  const float* W2     = (const float*)d_in[9];
  const float* b2     = (const float*)d_in[10];
  const float* W3     = (const float*)d_in[11];
  const float* b3     = (const float*)d_in[12];
  const float* k_emb  = (const float*)d_in[13];

  // workspace carving (~631 MB; weights L2-resident, activations streamed)
  char* ws = (char*)d_ws;
  size_t off = 0;
  auto carve = [&](size_t bytes) -> void* {
    void* p = ws + off;
    off = (off + bytes + 255) & ~(size_t)255;
    return p;
  };
  bf16_t* w1n  = (bf16_t*)carve((size_t)FEAT_DIM * H_DIM * 2);   // W1 bf16  [416,2048]
  bf16_t* w1t  = (bf16_t*)carve((size_t)FEAT_DIM * H_DIM * 2);   // W1^T     [2048,416]
  bf16_t* w2n  = (bf16_t*)carve((size_t)H_DIM * H_DIM * 2);      // W2 bf16  [2048,2048]
  bf16_t* w2t  = (bf16_t*)carve((size_t)H_DIM * H_DIM * 2);      // W2^T
  bf16_t* feat = (bf16_t*)carve((size_t)B_SZ * FEAT_DIM * 2);
  float*  yhat = (float*) carve((size_t)B_SZ * YN * 4);
  float*  yneg = (float*) carve((size_t)B_SZ * YN * 4);
  float*  pre1 = (float*) carve((size_t)B_SZ * H_DIM * 4);
  float*  pre2 = (float*) carve((size_t)B_SZ * H_DIM * 4);
  bf16_t* h1   = (bf16_t*)carve((size_t)B_SZ * H_DIM * 2);
  bf16_t* h2   = (bf16_t*)carve((size_t)B_SZ * H_DIM * 2);
  bf16_t* dh2  = (bf16_t*)carve((size_t)B_SZ * H_DIM * 2);
  bf16_t* dh1  = (bf16_t*)carve((size_t)B_SZ * H_DIM * 2);
  float*  dft  = (float*) carve((size_t)B_SZ * FEAT_DIM * 4);
  float*  epos = (float*) carve((size_t)B_SZ * 4);
  float*  eneg = (float*) carve((size_t)B_SZ * 4);
  float*  acc  = (float*) carve(8 * 4);

  // GEMM grids: block tile 256(M) x 32(N), 8 waves/block, 32x32 per wave
  dim3 blk(256);
  dim3 gH (H_DIM / 32,    B_SZ / 256);   // N=2048
  dim3 gF (FEAT_DIM / 32, B_SZ / 256);   // N=416 (13 tiles)

  auto fwd = [&](float* evec) {
    k_wmma_gemm<EPI_FWD><<<gH, blk, 0, stream>>>(feat, w1t, b1, nullptr, pre1, h1, nullptr,
                                                 B_SZ, FEAT_DIM, H_DIM);
    k_wmma_gemm<EPI_FWD><<<gH, blk, 0, stream>>>(h1, w2t, b2, nullptr, pre2, h2, nullptr,
                                                 B_SZ, H_DIM, H_DIM);
    if (evec) k_dot_w3<<<B_SZ / 8, 256, 0, stream>>>(h2, W3, b3, evec);
  };
  auto bwd = [&]() {
    k_dh2<<<2048, 256, 0, stream>>>(pre2, W3, dh2);
    k_wmma_gemm<EPI_DGELU><<<gH, blk, 0, stream>>>(dh2, w2n, nullptr, pre1, nullptr, dh1,
                                                   nullptr, B_SZ, H_DIM, H_DIM);
    k_wmma_gemm<EPI_PLAIN><<<gF, blk, 0, stream>>>(dh1, w1n, nullptr, nullptr, nullptr,
                                                   nullptr, dft, B_SZ, H_DIM, FEAT_DIM);
  };

  k_zero<<<1, 32, 0, stream>>>(acc);
  k_cvt<<<2048, 256, 0, stream>>>(W1, w1n, w1t, FEAT_DIM, H_DIM);
  k_cvt<<<4096, 256, 0, stream>>>(W2, w2n, w2t, H_DIM, H_DIM);

  // ---- positive branch: e_pos + score_pred + mse ----
  k_noise<<<(B_SZ * CELLS + 255) / 256, 256, 0, stream>>>(y_true, eps_p, k_idx, yhat);
  k_feat<<<4096, 256, 0, stream>>>(yhat, cond, k_idx, k_emb, feat);
  fwd(epos);
  bwd();
  k_mse<<<2048, 256, 0, stream>>>(dft, yhat, y_true, k_idx, acc);

  // ---- negative synthesis + Langevin refinement ----
  k_make_neg<<<(B_SZ + 255) / 256, 256, 0, stream>>>(y_true, sel, rv, yneg);
  for (int s = 0; s < REFINE_STEPS; ++s) {
    k_feat<<<4096, 256, 0, stream>>>(yneg, cond, k_idx, k_emb, feat);
    fwd(nullptr);   // energy value not needed, only dE/dy
    bwd();
    k_refine<<<2048, 256, 0, stream>>>(dft, yneg);
  }

  // ---- negative branch: e_neg ----
  k_noise<<<(B_SZ * CELLS + 255) / 256, 256, 0, stream>>>(yneg, eps_n, k_idx, yhat);
  k_feat<<<4096, 256, 0, stream>>>(yhat, cond, k_idx, k_emb, feat);
  fwd(eneg);

  // ---- losses ----
  k_ereduce<<<64, 256, 0, stream>>>(epos, eneg, acc);
  k_final<<<1, 32, 0, stream>>>(acc, (float*)d_out);
}
